// TrajectoryGenerator_11184094839490
// MI455X (gfx1250) — matrix operand order repaired and checked by reading, not compiled, not run on hardware
//
#include <hip/hip_runtime.h>
#include <hip/hip_bf16.h>

typedef _Float16 f16;
typedef __attribute__((ext_vector_type(16))) _Float16 v16h;
typedef __attribute__((ext_vector_type(8)))  _Float16 v8h;
typedef __attribute__((ext_vector_type(8)))  float    v8f;

#define NSCENE 256
#define NPED   16
#define KIN    128    // E + H
#define PRE1   512
#define BNK    1024
#define XSTR   136    // padded LDS row stride (halves): 272B = 17*16B
#define Y1STR  520    // padded LDS row stride (halves): 1040B = 65*16B
#define RCHUNK 128    // rows per chunk (8 i-groups); 2 chunks per scene
#define LDS_X_BYTES  (RCHUNK * XSTR * 2)            // 34816
#define LDS_Y1_BYTES (RCHUNK * Y1STR * 2)           // 133120
#define LDS_TOTAL    (LDS_X_BYTES + LDS_Y1_BYTES)   // 167936 <= 320KB/WGP

// ---------------------------------------------------------------------------
// Prep: transpose Wp1/Wp2 to column-major f16 (each WMMA B-fragment lane load
// becomes 32 contiguous bytes) and fold BatchNorm + linear bias into
// per-channel scale/shift:  y = relu(mat*scale + shift).
// ---------------------------------------------------------------------------
__global__ __launch_bounds__(256) void prep_kernel(
    const float* __restrict__ Wp1, const float* __restrict__ Wp2,
    const float* __restrict__ bp1, const float* __restrict__ gp1,
    const float* __restrict__ btp1, const float* __restrict__ mp1,
    const float* __restrict__ vp1,
    const float* __restrict__ bp2, const float* __restrict__ gp2,
    const float* __restrict__ btp2, const float* __restrict__ mp2,
    const float* __restrict__ vp2,
    f16* __restrict__ wt1, f16* __restrict__ wt2,
    float* __restrict__ s1, float* __restrict__ h1,
    float* __restrict__ s2, float* __restrict__ h2)
{
  const int idx = blockIdx.x * 256 + threadIdx.x;
  const int N1 = KIN * PRE1;          // 65536
  const int N2 = PRE1 * BNK;          // 524288
  if (idx < N1) {
    int col = idx >> 7, k = idx & (KIN - 1);
    wt1[col * KIN + k] = (f16)Wp1[k * PRE1 + col];
  } else if (idx < N1 + N2) {
    int r = idx - N1;
    int col = r >> 9, k = r & (PRE1 - 1);
    wt2[col * PRE1 + k] = (f16)Wp2[k * BNK + col];
  } else if (idx < N1 + N2 + PRE1) {
    int c = idx - (N1 + N2);
    float sc = gp1[c] * rsqrtf(vp1[c] + 1e-5f);
    s1[c] = sc;
    h1[c] = (bp1[c] - mp1[c]) * sc + btp1[c];
  } else if (idx < N1 + N2 + PRE1 + BNK) {
    int c = idx - (N1 + N2 + PRE1);
    float sc = gp2[c] * rsqrtf(vp2[c] + 1e-5f);
    s2[c] = sc;
    h2[c] = (bp2[c] - mp2[c]) * sc + btp2[c];
  }
}

// ---------------------------------------------------------------------------
// Fused per-scene kernel (softmax over size-1 axis == 1 -> attention branch
// is dead).  1 block = 1 scene, 256 threads = 8 wave32s, 2 chunks of 128 rows.
// Column tiles are processed in PAIRS with 8 row-tiles of accumulators live,
// so each B fragment feeds 8 rt and each A fragment feeds 2 WMMAs.
// ---------------------------------------------------------------------------
__global__ __launch_bounds__(256) void fused_scene_kernel(
    const float* __restrict__ h_states,   // [B,H]
    const float* __restrict__ end_pos,    // [B,2]
    const float* __restrict__ W_sp,       // [2,64]
    const float* __restrict__ b_sp,       // [64]
    const f16*  __restrict__ wt1,         // [512][128] col-major f16
    const f16*  __restrict__ wt2,         // [1024][512] col-major f16
    const float* __restrict__ s1, const float* __restrict__ h1,
    const float* __restrict__ s2, const float* __restrict__ h2,
    float* __restrict__ out)              // [4096][1024]
{
  extern __shared__ __align__(32) char smem_raw[];
  f16* sX  = (f16*)smem_raw;                      // [RCHUNK][XSTR]
  f16* sY1 = (f16*)(smem_raw + LDS_X_BYTES);      // [RCHUNK][Y1STR]

  const int s    = blockIdx.x;
  const int t    = threadIdx.x;
  const int w    = t >> 5;       // wave id 0..7
  const int lane = t & 31;
  const int lo   = lane & 15;    // N index within tile
  const int hi   = lane >> 4;    // K/M half select

  for (int ch = 0; ch < 2; ++ch) {
    // ---- Phase A: build X chunk (128 rows x 128 cols, f16) ----------------
    {
      const int rr   = t >> 1;                   // 0..127
      const int cb   = (t & 1) << 6;             // 0 or 64
      const int srow = ch * RCHUNK + rr;         // i*16 + j within scene
      const int i    = srow >> 4, j = srow & 15;
      const float pxi = end_pos[(s * NPED + i) * 2 + 0];
      const float pyi = end_pos[(s * NPED + i) * 2 + 1];
      const float pxj = end_pos[(s * NPED + j) * 2 + 0];
      const float pyj = end_pos[(s * NPED + j) * 2 + 1];
      const float rx = pxj - pxi, ry = pyj - pyi;   // rel = pos[j]-pos[i]
      if (cb == 0) {
        #pragma unroll 8
        for (int e = 0; e < 64; ++e)
          sX[rr * XSTR + e] = (f16)(0.05f * (rx * W_sp[e] + ry * W_sp[64 + e] + b_sp[e]));
      } else {
        const float* hrow = h_states + (s * NPED + j) * 64;
        #pragma unroll 8
        for (int e = 0; e < 64; ++e)
          sX[rr * XSTR + 64 + e] = (f16)(0.05f * hrow[e]);
      }
    }
    __syncthreads();

    // ---- Phase B: GEMM1 [128,128]@[128,512] + BN + ReLU -> sY1 ------------
    // 32 col-tiles / 8 waves = 4 per wave, processed as 2 pairs.
    for (int qp = 0; qp < 2; ++qp) {
      const int n0a = (w * 4 + qp * 2) * 16;
      const int n0b = n0a + 16;
      v8f accA[8] = {}, accB[8] = {};
      for (int k0 = 0; k0 < KIN / 32; ++k0) {
        v16h bA = *(const v16h*)(wt1 + (n0a + lo) * KIN + k0 * 32 + hi * 16);
        v16h bB = *(const v16h*)(wt1 + (n0b + lo) * KIN + k0 * 32 + hi * 16);
        #pragma unroll
        for (int rt = 0; rt < 8; ++rt) {
          const f16* p = sX + (rt * 16 + lo) * XSTR + k0 * 32 + hi * 16;
          v16h a;
          *(v8h*)&a       = *(const v8h*)p;
          *((v8h*)&a + 1) = *(const v8h*)(p + 8);
          accA[rt] = __builtin_amdgcn_wmma_f32_16x16x32_f16(false, a, false, bA,
                                                            (short)0, accA[rt], false, false);
          accB[rt] = __builtin_amdgcn_wmma_f32_16x16x32_f16(false, a, false, bB,
                                                            (short)0, accB[rt], false, false);
        }
      }
      const float lsA = s1[n0a + lo], lhA = h1[n0a + lo];
      const float lsB = s1[n0b + lo], lhB = h1[n0b + lo];
      #pragma unroll
      for (int rt = 0; rt < 8; ++rt) {
        #pragma unroll
        for (int r = 0; r < 8; ++r) {
          const int row = rt * 16 + r + 8 * hi;
          sY1[row * Y1STR + n0a + lo] = (f16)fmaxf(accA[rt][r] * lsA + lhA, 0.f);
          sY1[row * Y1STR + n0b + lo] = (f16)fmaxf(accB[rt][r] * lsB + lhB, 0.f);
        }
      }
    }
    __syncthreads();

    // ---- Phase C: GEMM2 [128,512]@[512,1024] + BN + ReLU + max over j -----
    // 64 col-tiles / 8 waves = 8 per wave, processed as 4 pairs.
    for (int qp = 0; qp < 4; ++qp) {
      const int n0a = (w * 8 + qp * 2) * 16;
      const int n0b = n0a + 16;
      v8f accA[8] = {}, accB[8] = {};
      for (int k0 = 0; k0 < PRE1 / 32; ++k0) {
        v16h bA = *(const v16h*)(wt2 + (n0a + lo) * PRE1 + k0 * 32 + hi * 16);
        v16h bB = *(const v16h*)(wt2 + (n0b + lo) * PRE1 + k0 * 32 + hi * 16);
        #pragma unroll
        for (int rt = 0; rt < 8; ++rt) {
          const f16* p = sY1 + (rt * 16 + lo) * Y1STR + k0 * 32 + hi * 16;
          v16h a;
          *(v8h*)&a       = *(const v8h*)p;
          *((v8h*)&a + 1) = *(const v8h*)(p + 8);
          accA[rt] = __builtin_amdgcn_wmma_f32_16x16x32_f16(false, a, false, bA,
                                                            (short)0, accA[rt], false, false);
          accB[rt] = __builtin_amdgcn_wmma_f32_16x16x32_f16(false, a, false, bB,
                                                            (short)0, accB[rt], false, false);
        }
      }
      const float lsA = s2[n0a + lo], lhA = h2[n0a + lo];
      const float lsB = s2[n0b + lo], lhB = h2[n0b + lo];
      #pragma unroll
      for (int rt = 0; rt < 8; ++rt) {
        float mA = 0.f, mB = 0.f;                 // relu outputs >= 0
        #pragma unroll
        for (int r = 0; r < 8; ++r) {
          mA = fmaxf(mA, fmaxf(accA[rt][r] * lsA + lhA, 0.f));
          mB = fmaxf(mB, fmaxf(accB[rt][r] * lsB + lhB, 0.f));
        }
        // combine M=0..7 half (lanes 0-15) with M=8..15 half (lanes 16-31)
        mA = fmaxf(mA, __shfl_xor(mA, 16, 32));
        mB = fmaxf(mB, __shfl_xor(mB, 16, 32));
        if (hi == 0) {
          const int ig = ch * 8 + rt;             // pooled output row s*16+ig
          out[(s * NPED + ig) * BNK + n0a + lo] = mA;
          out[(s * NPED + ig) * BNK + n0b + lo] = mB;
        }
      }
    }
    __syncthreads();
  }
}

extern "C" void kernel_launch(void* const* d_in, const int* in_sizes, int n_in,
                              void* d_out, int out_size, void* d_ws, size_t ws_size,
                              hipStream_t stream) {
  (void)in_sizes; (void)n_in; (void)out_size; (void)ws_size;
  const float* h_states = (const float*)d_in[0];
  const float* end_pos  = (const float*)d_in[1];
  // d_in[2] vx, d_in[3] vy, d_in[6..19]: dead (softmax over size-1 axis == 1)
  const float* W_sp = (const float*)d_in[4];
  const float* b_sp = (const float*)d_in[5];
  const float* Wp1  = (const float*)d_in[20];
  const float* bp1  = (const float*)d_in[21];
  const float* gp1  = (const float*)d_in[22];
  const float* btp1 = (const float*)d_in[23];
  const float* mp1  = (const float*)d_in[24];
  const float* vp1  = (const float*)d_in[25];
  const float* Wp2  = (const float*)d_in[26];
  const float* bp2  = (const float*)d_in[27];
  const float* gp2  = (const float*)d_in[28];
  const float* btp2 = (const float*)d_in[29];
  const float* mp2  = (const float*)d_in[30];
  const float* vp2  = (const float*)d_in[31];

  // workspace layout (bytes)
  char* ws = (char*)d_ws;
  f16*   wt1 = (f16*)(ws + 0);                      // 512*128*2  = 131072
  f16*   wt2 = (f16*)(ws + 131072);                 // 1024*512*2 = 1048576
  float* s1  = (float*)(ws + 1179648);              // 512*4
  float* h1  = (float*)(ws + 1181696);              // 512*4
  float* s2  = (float*)(ws + 1183744);              // 1024*4
  float* h2  = (float*)(ws + 1187840);              // 1024*4

  const int prepN = 128 * 512 + 512 * 1024 + 512 + 1024;   // 591360
  prep_kernel<<<(prepN + 255) / 256, 256, 0, stream>>>(
      Wp1, Wp2, bp1, gp1, btp1, mp1, vp1, bp2, gp2, btp2, mp2, vp2,
      wt1, wt2, s1, h1, s2, h2);

  // 164KB dynamic LDS (<= 320KB/WGP on CDNA5); host-side attribute call is
  // not a stream op, so it is graph-capture safe and deterministic.
  (void)hipFuncSetAttribute((const void*)fused_scene_kernel,
                            hipFuncAttributeMaxDynamicSharedMemorySize,
                            LDS_TOTAL);
  fused_scene_kernel<<<NSCENE, 256, LDS_TOTAL, stream>>>(
      h_states, end_pos, W_sp, b_sp, wt1, wt2, s1, h1, s2, h2, (float*)d_out);
}
